// ECABottleneck_40931038331270
// MI455X (gfx1250) — matrix operand order repaired and checked by reading, not compiled, not run on hardware
//
#include <hip/hip_runtime.h>

typedef __attribute__((ext_vector_type(16))) _Float16 v16h;
typedef __attribute__((ext_vector_type(8)))  _Float16 v8h;
typedef __attribute__((ext_vector_type(8)))  float    v8f;
typedef __attribute__((ext_vector_type(4)))  unsigned int v4u;
typedef __attribute__((ext_vector_type(8)))  int          v8i;
typedef __attribute__((ext_vector_type(4)))  int          v4i;

// Problem geometry (fixed by the reference)
#define NPIX 3136        // 56*56
#define HPAD 58          // padded spatial
#define BATCH 16

#ifndef __has_builtin
#define __has_builtin(x) 0
#endif
#if __has_builtin(__builtin_amdgcn_tensor_load_to_lds) && __has_builtin(__builtin_amdgcn_s_wait_tensorcnt)
#define HAVE_TDM 1
#else
#define HAVE_TDM 0
#endif

// ---------------------------------------------------------------------------
// TDM helpers: 2D tile load (BM=128 rows x 64 halves, row stride in elements)
// into LDS with pad: 32 dwords data then 4 dwords pad -> LDS row = 72 halves.
// Descriptor layout per CDNA5 ISA.
// ---------------------------------------------------------------------------
__device__ __forceinline__ void tdm_load_tile_2d(unsigned lds_addr, const void* gsrc,
                                                 int tensor_rows, int row_stride_elems)
{
#if HAVE_TDM
    unsigned long long ga = (unsigned long long)(uintptr_t)gsrc;
    // group0: count=1 | lds_addr | global_addr[56:0] | type=2 ("image")
    v4u g0 = { 1u,
               lds_addr,
               (unsigned)ga,
               (unsigned)((ga >> 32) & 0x01ffffffu) | (2u << 30) };
    // group1: wg_mask=0, data_size=1(2B), pad_enable, pad_interval=4 (32 dwords),
    //         pad_amount=3 (4 dwords); tensor_dim0=64, tensor_dim1=tensor_rows,
    //         tile_dim0=64, tile_dim1=128, dim0_stride=row_stride_elems
    v8i g1 = { (int)((1u << 16) | (1u << 20) | (4u << 22) | (3u << 25)),
               (int)(64u << 16),                                   // tensor_dim0 lo16
               (int)(((unsigned)tensor_rows & 0xffffu) << 16),     // tensor_dim1 lo16
               (int)(64u << 16),                                   // tile_dim0
               128,                                                // tile_dim1
               row_stride_elems,                                   // dim0_stride lo32
               0, 0 };
    v4i z4 = { 0, 0, 0, 0 };
#if __clang_major__ >= 23
    v8i z8 = { 0, 0, 0, 0, 0, 0, 0, 0 };
    __builtin_amdgcn_tensor_load_to_lds(g0, g1, z4, z4, z8, 0);
#else
    __builtin_amdgcn_tensor_load_to_lds(g0, g1, z4, z4, 0);
#endif
#else
    (void)lds_addr; (void)gsrc; (void)tensor_rows; (void)row_stride_elems;
#endif
}

__device__ __forceinline__ void tdm_wait_one()  {
#if HAVE_TDM
    __builtin_amdgcn_s_wait_tensorcnt(1);
#endif
}
__device__ __forceinline__ void tdm_wait_zero() {
#if HAVE_TDM
    __builtin_amdgcn_s_wait_tensorcnt(0);
#endif
}

// ---------------------------------------------------------------------------
// prep: fold BN params, convert weights to f16, pack w2 (O,I,3,3)->(tap,O,I)
// ---------------------------------------------------------------------------
__global__ __launch_bounds__(256) void prep_kernel(
    const float* __restrict__ w1, const float* __restrict__ w2, const float* __restrict__ w3,
    const float* __restrict__ g1, const float* __restrict__ b1, const float* __restrict__ m1, const float* __restrict__ v1,
    const float* __restrict__ g2, const float* __restrict__ b2, const float* __restrict__ m2, const float* __restrict__ v2,
    const float* __restrict__ g3, const float* __restrict__ b3, const float* __restrict__ m3, const float* __restrict__ v3,
    _Float16* __restrict__ W1B, _Float16* __restrict__ W2P, _Float16* __restrict__ W3B,
    float* __restrict__ S1, float* __restrict__ B1,
    float* __restrict__ S2, float* __restrict__ B2,
    float* __restrict__ S3, float* __restrict__ B3)
{
    const long N1 = 256L * 1024, N2 = 256L * 256 * 9, N3 = 1024L * 256;
    const long total = N1 + N2 + N3 + 256 + 256 + 1024;
    for (long i = blockIdx.x * 256L + threadIdx.x; i < total; i += (long)gridDim.x * 256L) {
        long j = i;
        if (j < N1) { W1B[j] = (_Float16)w1[j]; continue; }
        j -= N1;
        if (j < N2) {
            // src flat index j = (co*256 + ci)*9 + t ; dst = (t*256 + co)*256 + ci
            int co = (int)(j / 2304); int r = (int)(j % 2304); int ci = r / 9; int t = r % 9;
            W2P[((long)t * 256 + co) * 256 + ci] = (_Float16)w2[j];
            continue;
        }
        j -= N2;
        if (j < N3) { W3B[j] = (_Float16)w3[j]; continue; }
        j -= N3;
        if (j < 256) { float s = g1[j] * rsqrtf(v1[j] + 1e-5f); S1[j] = s; B1[j] = b1[j] - m1[j] * s; continue; }
        j -= 256;
        if (j < 256) { float s = g2[j] * rsqrtf(v2[j] + 1e-5f); S2[j] = s; B2[j] = b2[j] - m2[j] * s; continue; }
        j -= 256;
        {  float s = g3[j] * rsqrtf(v3[j] + 1e-5f); S3[j] = s; B3[j] = b3[j] - m3[j] * s; }
    }
}

// ---------------------------------------------------------------------------
// Transpose-convert x: (16,1024,56,56) f32 NCHW -> (16,3136,1024) f16 NHWC
// ---------------------------------------------------------------------------
__global__ __launch_bounds__(256) void transpose_cvt_kernel(
    const float* __restrict__ x, _Float16* __restrict__ xt)
{
    __shared__ float tile[32][33];
    const int pb = blockIdx.x * 32;   // pixel base (98 blocks)
    const int cb = blockIdx.y * 32;   // channel base (32 blocks)
    const int img = blockIdx.z;
    const int tp = threadIdx.x & 31;
    const int tr = threadIdx.x >> 5;  // 0..7
#pragma unroll
    for (int i = 0; i < 4; ++i) {
        int c = tr + i * 8;
        tile[c][tp] = x[((size_t)img * 1024 + cb + c) * NPIX + pb + tp];
    }
    __syncthreads();
#pragma unroll
    for (int i = 0; i < 4; ++i) {
        int p = tr + i * 8;
        xt[((size_t)img * NPIX + pb + p) * 1024 + cb + tp] = (_Float16)tile[tp][p];
    }
}

// ---------------------------------------------------------------------------
// zero fill (dword granularity) for the padded activation buffers
// ---------------------------------------------------------------------------
__global__ __launch_bounds__(256) void zero_kernel(unsigned int* __restrict__ p, long nwords)
{
    for (long i = blockIdx.x * 256L + threadIdx.x; i < nwords; i += (long)gridDim.x * 256L)
        p[i] = 0u;
}

// ---------------------------------------------------------------------------
// WMMA implicit-GEMM conv kernel.
//   D[p, co] = sum_{tap, ci} A[pix(p,tap), ci] * W[tap, co, ci]
//   KD: K per tap (= channel stride of A), ND: out channels, TAPS: 1 or 9,
//   APAD: A is padded 58x58 NHWC, MODE 0: BN+ReLU->f16 padded NHWC,
//   MODE 1: raw f32 NCHW. USE_TDM: stage A tiles with tensor_load_to_lds
//   (double-buffered; flat A source only, i.e. TAPS==1 && APAD==0).
// Block tile 128(M pixels) x 128(N couts), 8 waves = 4(M) x 2(N),
// each wave: 2x4 tiles of v_wmma_f32_16x16x32_f16, K staged via LDS (BK=64).
// ---------------------------------------------------------------------------
template<int KD, int ND, int TAPS, int APAD, int MODE, int USE_TDM>
__global__ __launch_bounds__(256) void wmma_conv_kernel(
    const _Float16* __restrict__ A, const _Float16* __restrict__ W,
    const float* __restrict__ scale, const float* __restrict__ bias,
    _Float16* __restrict__ outH, float* __restrict__ outF)
{
    constexpr int BM = 128, BN = 128, BK = 64;
    constexpr int LDA = BK + 8;                 // 72 halves -> 144B rows (16B aligned)
    constexpr int kTdm = (USE_TDM && HAVE_TDM) ? 1 : 0;
    constexpr int NBUF = kTdm ? 2 : 1;
    __shared__ __align__(16) _Float16 As[NBUF * BM * LDA];

    const int tid  = threadIdx.x;
    const int lane = tid & 31;
    const int wave = tid >> 5;
    const int wm   = wave & 3;                  // 0..3  (M waves)
    const int wn   = wave >> 2;                 // 0..1  (N waves)
    const int img  = blockIdx.z;
    const int m0   = blockIdx.y * BM;           // pixel base
    const int n0   = blockIdx.x * BN;           // cout base
    const int lr   = lane & 15;
    const int hs   = lane >> 4;                 // lane half-select (per WMMA layouts)

    v8f acc[2][4] = {};

    // consume one staged A tile (two K=32 WMMA steps) against weights Wt at k0
    auto consume = [&](const _Float16* tile, const _Float16* Wt, int k0) {
#pragma unroll
        for (int kk = 0; kk < BK; kk += 32) {
            v16h a[2], b[4];
#pragma unroll
            for (int mi = 0; mi < 2; ++mi) {
                // A 16x32 f16 layout: lane<16: K0-7,K16-23 ; lane>=16: K8-15,K24-31
                int m = wm * 32 + mi * 16 + lr;
                const _Float16* base = &tile[m * LDA + kk + hs * 8];
                v8h lo = *(const v8h*)base;
                v8h hi = *(const v8h*)(base + 16);
                a[mi] = __builtin_shufflevector(lo, hi, 0,1,2,3,4,5,6,7,8,9,10,11,12,13,14,15);
            }
#pragma unroll
            for (int ni = 0; ni < 4; ++ni) {
                // B 32x16 f16 layout: lane<16: col n, K0-7,K16-23 ; lane>=16: col n, K8-15,K24-31
                int n = n0 + wn * 64 + ni * 16 + lr;
                const _Float16* wb = Wt + (size_t)n * KD + k0 + kk + hs * 8;
                v8h lo = *(const v8h*)wb;
                v8h hi = *(const v8h*)(wb + 16);
                b[ni] = __builtin_shufflevector(lo, hi, 0,1,2,3,4,5,6,7,8,9,10,11,12,13,14,15);
            }
#pragma unroll
            for (int mi = 0; mi < 2; ++mi)
#pragma unroll
                for (int ni = 0; ni < 4; ++ni)
                    acc[mi][ni] = __builtin_amdgcn_wmma_f32_16x16x32_f16(
                        false, a[mi], false, b[ni], (short)0, acc[mi][ni], false, false);
        }
    };

    if constexpr (kTdm) {
        // ---- async-tensor staging, double buffered, wave0 drives the TDM ----
        constexpr int NITER = KD / BK;
        const _Float16* Abase = A + ((size_t)img * NPIX + m0) * KD;
        const int rows = NPIX - m0;             // OOB rows -> hardware zero fill
        const bool leader = (__builtin_amdgcn_readfirstlane((int)(threadIdx.x >> 5)) == 0);
        const unsigned lds0 = (unsigned)(uintptr_t)(void*)&As[0];
        const unsigned lds1 = (unsigned)(uintptr_t)(void*)&As[BM * LDA];
        if (leader)
            tdm_load_tile_2d(lds0, Abase, rows, KD);
        for (int it = 0; it < NITER; ++it) {
            if (leader) {
                if (it + 1 < NITER) {
                    tdm_load_tile_2d((it & 1) ? lds0 : lds1,
                                     Abase + (size_t)(it + 1) * BK, rows, KD);
                    tdm_wait_one();             // tile 'it' complete (in-order TDM)
                } else {
                    tdm_wait_zero();
                }
            }
            __syncthreads();                    // release consumers
            consume(&As[(it & 1) * (BM * LDA)], W, it * BK);
            __syncthreads();                    // done reading before next overwrite
        }
    } else {
        // ---- vector-load staging (handles padded/shifted A addressing) ----
        for (int t = 0; t < TAPS; ++t) {
            const int dy = (TAPS == 9) ? (t / 3) : 1;
            const int dx = (TAPS == 9) ? (t % 3) : 1;
            const _Float16* Wt = W + (size_t)t * ND * KD;
            for (int k0 = 0; k0 < KD; k0 += BK) {
#pragma unroll
                for (int i = 0; i < 4; ++i) {
                    int c  = tid + i * 256;     // 1024 chunks of 8 halves
                    int m  = c >> 3;            // BK/8 = 8 chunks per row
                    int kc = (c & 7) * 8;
                    int p  = m0 + m;
                    v8h val = {};
                    if (p < NPIX) {
                        const _Float16* src;
                        if (APAD) {
                            int y = p / 56, xx = p - y * 56;
                            src = A + (((size_t)img * HPAD + (y + dy)) * HPAD + (xx + dx)) * KD + k0 + kc;
                        } else {
                            src = A + ((size_t)img * NPIX + p) * KD + k0 + kc;
                        }
                        val = *(const v8h*)src;
                    }
                    *(v8h*)&As[m * LDA + kc] = val;
                }
                __syncthreads();
                consume(&As[0], Wt, k0);
                __syncthreads();
            }
        }
    }

    // ---- epilogue: C/D layout lane = one cout column, 8 pixel rows ----
#pragma unroll
    for (int mi = 0; mi < 2; ++mi) {
#pragma unroll
        for (int ni = 0; ni < 4; ++ni) {
            int n = n0 + wn * 64 + ni * 16 + lr;   // cout for this lane
            float s = 1.f, bb = 0.f;
            if (MODE == 0) { s = scale[n]; bb = bias[n]; }
#pragma unroll
            for (int r = 0; r < 8; ++r) {
                int p = m0 + wm * 32 + mi * 16 + hs * 8 + r;
                if (p < NPIX) {
                    float v = acc[mi][ni][r];
                    if (MODE == 0) {
                        v = v * s + bb;
                        v = v > 0.f ? v : 0.f;
                        int y = p / 56, xx = p - y * 56;
                        outH[(((size_t)img * HPAD + (y + 1)) * HPAD + (xx + 1)) * ND + n] = (_Float16)v;
                    } else {
                        outF[((size_t)img * ND + n) * NPIX + p] = v;
                    }
                }
            }
        }
    }
}

// ---------------------------------------------------------------------------
// ECA: max over center 2x2 crop -> conv1d(k=5, pad 2) over channels -> sigmoid
// ---------------------------------------------------------------------------
__global__ __launch_bounds__(256) void eca_kernel(
    const float* __restrict__ out, const float* __restrict__ we, float* __restrict__ sig)
{
    __shared__ float sq[1024];
    const int b = blockIdx.x;
#pragma unroll
    for (int j = 0; j < 4; ++j) {
        int c = threadIdx.x + j * 256;
        const float* p = out + ((size_t)b * 1024 + c) * NPIX;
        float m = fmaxf(fmaxf(p[27 * 56 + 27], p[27 * 56 + 28]),
                        fmaxf(p[28 * 56 + 27], p[28 * 56 + 28]));
        sq[c] = m;
    }
    __syncthreads();
#pragma unroll
    for (int j = 0; j < 4; ++j) {
        int c = threadIdx.x + j * 256;
        float y = 0.f;
#pragma unroll
        for (int i = 0; i < 5; ++i) {
            int cc = c + i - 2;
            if (cc >= 0 && cc < 1024) y += sq[cc] * we[i];
        }
        sig[b * 1024 + c] = 1.0f / (1.0f + __expf(-y));
    }
}

// ---------------------------------------------------------------------------
// final: out = relu(bn3(out) * sig + x), in place on d_out (fp32, float4)
// ---------------------------------------------------------------------------
__global__ __launch_bounds__(256) void final_kernel(
    float* __restrict__ out, const float* __restrict__ x,
    const float* __restrict__ S3, const float* __restrict__ B3,
    const float* __restrict__ sig)
{
    const long n4 = (long)BATCH * 1024 * (NPIX / 4);
    for (long e = blockIdx.x * 256L + threadIdx.x; e < n4; e += (long)gridDim.x * 256L) {
        int bc = (int)(e / 784);             // b*1024 + c
        int c  = bc & 1023;
        float4 o  = ((const float4*)out)[e];
        float4 xr = ((const float4*)x)[e];
        float  s  = S3[c], bb = B3[c], sg = sig[bc];
        float4 r;
        r.x = fmaxf((o.x * s + bb) * sg + xr.x, 0.f);
        r.y = fmaxf((o.y * s + bb) * sg + xr.y, 0.f);
        r.z = fmaxf((o.z * s + bb) * sg + xr.z, 0.f);
        r.w = fmaxf((o.w * s + bb) * sg + xr.w, 0.f);
        ((float4*)out)[e] = r;
    }
}

// ---------------------------------------------------------------------------
extern "C" void kernel_launch(void* const* d_in, const int* in_sizes, int n_in,
                              void* d_out, int out_size, void* d_ws, size_t ws_size,
                              hipStream_t stream)
{
    const float* x  = (const float*)d_in[0];
    const float* w1 = (const float*)d_in[1];
    const float* g1 = (const float*)d_in[2];
    const float* b1 = (const float*)d_in[3];
    const float* m1 = (const float*)d_in[4];
    const float* v1 = (const float*)d_in[5];
    const float* w2 = (const float*)d_in[6];
    const float* g2 = (const float*)d_in[7];
    const float* b2 = (const float*)d_in[8];
    const float* m2 = (const float*)d_in[9];
    const float* v2 = (const float*)d_in[10];
    const float* w3 = (const float*)d_in[11];
    const float* g3 = (const float*)d_in[12];
    const float* b3 = (const float*)d_in[13];
    const float* m3 = (const float*)d_in[14];
    const float* v3 = (const float*)d_in[15];
    const float* we = (const float*)d_in[16];
    float* out = (float*)d_out;

    // workspace layout
    char* ws = (char*)d_ws;
    size_t off = 0;
    auto take = [&](size_t bytes) { char* p = ws + off; off = (off + bytes + 255) & ~(size_t)255; return p; };

    const size_t XT_BYTES  = (size_t)BATCH * NPIX * 1024 * 2;       // 102.8 MB f16 NHWC
    const size_t YP_BYTES  = (size_t)BATCH * HPAD * HPAD * 256 * 2; // 27.6 MB padded NHWC f16

    _Float16* XT  = (_Float16*)take(XT_BYTES);
    char*     YP  = take(2 * YP_BYTES);          // Y1P and Y2P contiguous (one zero pass)
    _Float16* Y1P = (_Float16*)YP;
    _Float16* Y2P = (_Float16*)(YP + YP_BYTES);
    _Float16* W1B = (_Float16*)take(256 * 1024 * 2);
    _Float16* W2P = (_Float16*)take(9 * 256 * 256 * 2);
    _Float16* W3B = (_Float16*)take(1024 * 256 * 2);
    float* S1 = (float*)take(256 * 4);
    float* B1 = (float*)take(256 * 4);
    float* S2 = (float*)take(256 * 4);
    float* B2 = (float*)take(256 * 4);
    float* S3 = (float*)take(1024 * 4);
    float* B3 = (float*)take(1024 * 4);
    float* SIG = (float*)take((size_t)BATCH * 1024 * 4);
    (void)ws_size; (void)in_sizes; (void)n_in; (void)out_size;

    // 1) fold BN, convert/pack weights
    prep_kernel<<<1024, 256, 0, stream>>>(w1, w2, w3,
        g1, b1, m1, v1, g2, b2, m2, v2, g3, b3, m3, v3,
        W1B, W2P, W3B, S1, B1, S2, B2, S3, B3);

    // 2) x NCHW f32 -> NHWC f16
    transpose_cvt_kernel<<<dim3(98, 32, BATCH), 256, 0, stream>>>(x, XT);

    // 3) zero padded activation buffers (both at once)
    zero_kernel<<<2048, 256, 0, stream>>>((unsigned int*)YP, (long)(2 * YP_BYTES / 4));

    // 4) conv1 1x1 (K=1024 -> N=256) + BN1 + ReLU -> Y1P  [TDM-staged A]
    wmma_conv_kernel<1024, 256, 1, 0, 0, 1><<<dim3(2, 25, BATCH), 256, 0, stream>>>(
        XT, W1B, S1, B1, Y1P, nullptr);

    // 5) conv2 3x3 (9 taps x K=256 -> N=256) + BN2 + ReLU -> Y2P
    wmma_conv_kernel<256, 256, 9, 1, 0, 0><<<dim3(2, 25, BATCH), 256, 0, stream>>>(
        Y1P, W2P, S2, B2, Y2P, nullptr);

    // 6) conv3 1x1 (K=256 -> N=1024), raw f32 NCHW straight into d_out
    wmma_conv_kernel<256, 1024, 1, 1, 1, 0><<<dim3(8, 25, BATCH), 256, 0, stream>>>(
        Y2P, W3B, nullptr, nullptr, nullptr, out);

    // 7) ECA squeeze (2x2 center crop max) + conv1d + sigmoid
    eca_kernel<<<BATCH, 256, 0, stream>>>(out, we, SIG);

    // 8) out = relu(bn3(out) * sig + residual), in place
    final_kernel<<<8192, 256, 0, stream>>>(out, x, S3, B3, SIG);
}
